// VSSM_H_29265907155740
// MI455X (gfx1250) — compile-verified, matrix operand order checked by reading, not run on hardware
//
#include <hip/hip_runtime.h>

// ---------------------------------------------------------------------------
// VMamba 2D selective-scan block for MI455X (gfx1250, wave32, WMMA).
// in_proj GEMM -> dwconv3x3+SiLU -> x_dbl GEMM (4-direction gather) ->
// dt GEMM + softplus -> CHUNKED selective scan (pass1 / chain / pass2) ->
// LN + SiLU gate -> out_proj GEMM.  Dense projections use
// v_wmma_f32_16x16x32_f16; each wave computes a 4-tile strip sharing the
// streaming operand fragment.
// ---------------------------------------------------------------------------

#define B0   4
#define Hh   64
#define Ww   64
#define LL   4096          // Hh*Ww
#define DM   192           // d_model
#define DI   384           // d_inner
#define DSN  16            // d_state
#define DTR  96            // dt_rank
#define KD   4             // directions
#define M1   16384         // B0*LL
#define NXZ  768           // 2*DI
#define CDBL 128           // DTR + 2*DSN
#define NCH  (B0*KD*DI)    // 6144 scan channels
#define NCK  16            // scan chunks
#define CS   (LL/NCK)      // 256 steps per chunk

typedef __attribute__((ext_vector_type(16))) _Float16 v16h;
typedef __attribute__((ext_vector_type(8)))  float    v8f;

// A-fragment (and B stored as [N][K] row-major): lane&15 selects row/col,
// lane>>4 selects K-half per ISA 16-bit operand layout.
__device__ __forceinline__ v16h frag_ld_rowmajor(const float* __restrict__ base,
                                                 int ld, int lane) {
  int row = lane & 15;
  int ko  = (lane >> 4) << 3;
  const float* p = base + (long)row * ld + ko;
  v16h f;
#pragma unroll
  for (int i = 0; i < 8; ++i) {
    f[i]     = (_Float16)p[i];
    f[8 + i] = (_Float16)p[16 + i];
  }
  return f;
}

// B-fragment for [K][N] storage: caller folds the per-lane column into
// baseLane; K index strides by kstride.
__device__ __forceinline__ v16h frag_ld_kstride(const float* __restrict__ baseLane,
                                                long kstride, int lane) {
  int ko = (lane >> 4) << 3;
  const float* p = baseLane + (long)ko * kstride;
  v16h f;
#pragma unroll
  for (int i = 0; i < 8; ++i) {
    f[i]     = (_Float16)p[(long)i * kstride];
    f[8 + i] = (_Float16)p[(long)(16 + i) * kstride];
  }
  return f;
}

// Spatial index for scan direction k at sequence position l.
// (gather index for u == scatter index for y, for all four directions)
__device__ __forceinline__ int smap(int k, int l) {
  switch (k & 3) {
    case 0:  return l;
    case 1:  return ((l & 63) << 6) | (l >> 6);
    case 2:  return (LL - 1) - l;
    default: { int lr = (LL - 1) - l; return ((lr & 63) << 6) | (lr >> 6); }
  }
}

// ---------------- K1: in_proj GEMM  xz[m][n] = sum_c x[m][c]*W[n][c] -------
// Wave strip: 16 x 64 (one A fragment shared by 4 WMMAs).
__global__ __launch_bounds__(256) void k_inproj(const float* __restrict__ x,
                                                const float* __restrict__ w,
                                                float* __restrict__ xc_pre,
                                                float* __restrict__ z) {
  int lane  = threadIdx.x & 31;
  int g     = blockIdx.x * 8 + (threadIdx.x >> 5);
  int mtile = (g / (NXZ / 64)) << 4;
  int nbase = (g % (NXZ / 64)) << 6;
  const float* pa = x + (long)mtile * DM;
  const float* pb = w + (long)nbase * DM;
  v8f acc[4] = {};
#pragma unroll
  for (int kk = 0; kk < DM; kk += 32) {
    __builtin_prefetch(pa + (long)(lane & 15) * DM + kk + 64, 0, 1);
    v16h a = frag_ld_rowmajor(pa + kk, DM, lane);
#pragma unroll
    for (int j = 0; j < 4; ++j) {
      v16h b = frag_ld_rowmajor(pb + (long)(j << 4) * DM + kk, DM, lane);
      acc[j] = __builtin_amdgcn_wmma_f32_16x16x32_f16(false, a, false, b,
                                                      (short)0, acc[j], false, false);
    }
  }
  int col = lane & 15;
  int rb  = (lane >> 4) << 3;
#pragma unroll
  for (int j = 0; j < 4; ++j) {
    int n = nbase + (j << 4) + col;
#pragma unroll
    for (int r = 0; r < 8; ++r) {
      int m  = mtile + rb + r;
      int bi = m >> 12, l = m & (LL - 1);
      if (n < DI) xc_pre[(((long)bi * DI + n) << 12) + l] = acc[j][r];
      else        z[(long)m * DI + (n - DI)] = acc[j][r];
    }
  }
}

// ---------------- K2: depthwise 3x3 conv (SAME) + SiLU ---------------------
__global__ __launch_bounds__(256) void k_conv(const float* __restrict__ xc_pre,
                                              const float* __restrict__ cw,
                                              const float* __restrict__ cb,
                                              float* __restrict__ xconv) {
  int idx = blockIdx.x * 256 + threadIdx.x;     // (b*DI+d)*LL + l
  int l   = idx & (LL - 1);
  int bd  = idx >> 12;
  int d   = bd % DI;
  int h   = l >> 6, w = l & 63;
  const float* src = xc_pre + (long)bd * LL;
  float acc = cb[d];
#pragma unroll
  for (int dy = -1; dy <= 1; ++dy) {
#pragma unroll
    for (int dx = -1; dx <= 1; ++dx) {
      int hh = h + dy, ww = w + dx;
      if (hh >= 0 && hh < Hh && ww >= 0 && ww < Ww)
        acc += src[(hh << 6) + ww] * cw[d * 9 + (dy + 1) * 3 + (dx + 1)];
    }
  }
  xconv[idx] = acc / (1.f + __expf(-acc));      // SiLU
}

// ---------------- K3: x_dbl[b,k,c,l] = sum_d xpw[k,c,d]*xs[b,k,d,l] --------
// Wave strip: 64 x 16 (one gathered B fragment shared by 4 WMMAs).
__global__ __launch_bounds__(256) void k_xdbl(const float* __restrict__ xconv,
                                              const float* __restrict__ xpw,
                                              float* __restrict__ xdbl) {
  int lane = threadIdx.x & 31;
  int g    = blockIdx.x * 8 + (threadIdx.x >> 5);
  const int grpPerBK = (CDBL / 64) * (LL / 16);      // 2*256
  int bk    = g / grpPerBK;
  int rem   = g % grpPerBK;
  int mbase = (rem >> 8) << 6;
  int ntile = (rem & 255) << 4;
  int k = bk & 3, b = bk >> 2;
  int scol = smap(k, ntile + (lane & 15));           // direction gather
  const float* pa     = xpw + (long)k * CDBL * DI;
  const float* pbLane = xconv + (long)b * DI * LL + scol;
  v8f acc[4] = {};
#pragma unroll
  for (int kk = 0; kk < DI; kk += 32) {
    v16h bfr = frag_ld_kstride(pbLane + (long)kk * LL, LL, lane);
#pragma unroll
    for (int j = 0; j < 4; ++j) {
      v16h a = frag_ld_rowmajor(pa + (long)(mbase + (j << 4)) * DI + kk, DI, lane);
      acc[j] = __builtin_amdgcn_wmma_f32_16x16x32_f16(false, a, false, bfr,
                                                      (short)0, acc[j], false, false);
    }
  }
  int col = lane & 15, rb = (lane >> 4) << 3;
#pragma unroll
  for (int j = 0; j < 4; ++j)
#pragma unroll
    for (int r = 0; r < 8; ++r) {
      int m = mbase + (j << 4) + rb + r;
      xdbl[(((long)bk * CDBL + m) << 12) + ntile + col] = acc[j][r];
    }
}

// ---------------- K4: delta = softplus(dtw[k] @ dts + bias), stored f16 ----
// Wave strip: 64 x 16 (one B fragment shared by 4 WMMAs).
__global__ __launch_bounds__(256) void k_delta(const float* __restrict__ xdbl,
                                               const float* __restrict__ dtw,
                                               const float* __restrict__ dtb,
                                               _Float16* __restrict__ delta) {
  int lane = threadIdx.x & 31;
  int g    = blockIdx.x * 8 + (threadIdx.x >> 5);
  const int grpPerBK = (DI / 64) * (LL / 16);        // 6*256
  int bk    = g / grpPerBK;
  int rem   = g % grpPerBK;
  int mbase = (rem >> 8) << 6;
  int ntile = (rem & 255) << 4;
  int k = bk & 3;
  const float* pa = dtw + (long)k * DI * DTR;
  const float* pb = xdbl + (long)bk * CDBL * LL + ntile;   // dts rows 0..95
  v8f acc[4] = {};
#pragma unroll
  for (int kk = 0; kk < DTR; kk += 32) {
    v16h bfr = frag_ld_kstride(pb + (long)kk * LL + (lane & 15), LL, lane);
#pragma unroll
    for (int j = 0; j < 4; ++j) {
      v16h a = frag_ld_rowmajor(pa + (long)(mbase + (j << 4)) * DTR + kk, DTR, lane);
      acc[j] = __builtin_amdgcn_wmma_f32_16x16x32_f16(false, a, false, bfr,
                                                      (short)0, acc[j], false, false);
    }
  }
  int col = lane & 15, rb = (lane >> 4) << 3;
#pragma unroll
  for (int j = 0; j < 4; ++j)
#pragma unroll
    for (int r = 0; r < 8; ++r) {
      int m = mbase + (j << 4) + rb + r;
      float v  = acc[j][r] + dtb[k * DI + m];
      float sp = (v > 20.f) ? v : log1pf(__expf(v));
      delta[(((long)bk * DI + m) << 12) + ntile + col] = (_Float16)sp;
    }
}

// ---------------- K5a: scan pass 1 — per-chunk F (zero-init) and sum(dl) ---
// grid = (B0*KD)*(DI/128)*NCK blocks of 128 threads; 1 thread = 1 channel.
__global__ __launch_bounds__(128) void k_scan1(const _Float16* __restrict__ delta,
                                               const float* __restrict__ xconv,
                                               const float* __restrict__ xdbl,
                                               const float* __restrict__ A_logs,
                                               float* __restrict__ Fbuf,
                                               float* __restrict__ Dsum) {
  int c     = blockIdx.x & (NCK - 1);
  int t     = blockIdx.x >> 4;
  int bk    = t / 3;
  int d     = (t % 3) * 128 + threadIdx.x;
  int k = bk & 3, b = bk >> 2;
  int gd = k * DI + d;
  float Ar[DSN];
#pragma unroll
  for (int n = 0; n < DSN; ++n) Ar[n] = -__expf(A_logs[gd * DSN + n]);
  const _Float16* dp = delta + (((long)bk * DI + d) << 12);
  const float*    up = xconv + (((long)b  * DI + d) << 12);
  const float*    bc = xdbl  + (((long)bk * CDBL + DTR) << 12);  // B rows
  __shared__ float sB[DSN];
  float h[DSN];
#pragma unroll
  for (int n = 0; n < DSN; ++n) h[n] = 0.f;
  float ds = 0.f;
  for (int l = c * CS; l < (c + 1) * CS; ++l) {
    if (threadIdx.x < DSN) sB[threadIdx.x] = bc[((long)threadIdx.x << 12) + l];
    __syncthreads();
    float dl = (float)dp[l];
    float du = dl * up[smap(k, l)];
    ds += dl;
#pragma unroll
    for (int n = 0; n < DSN; ++n)
      h[n] = __expf(dl * Ar[n]) * h[n] + du * sB[n];
    __syncthreads();
  }
  long ch = (long)bk * DI + d;
#pragma unroll
  for (int n = 0; n < DSN; ++n) Fbuf[(ch * NCK + c) * DSN + n] = h[n];
  Dsum[ch * NCK + c] = ds;
}

// ---------------- K5b: chain initial states across chunks ------------------
__global__ __launch_bounds__(256) void k_chain(const float* __restrict__ Fbuf,
                                               const float* __restrict__ Dsum,
                                               const float* __restrict__ A_logs,
                                               float* __restrict__ Hin) {
  int ch = blockIdx.x * 256 + threadIdx.x;    // 0..NCH-1
  int bk = ch / DI, d = ch % DI;
  int gd = (bk & 3) * DI + d;
  float Ar[DSN];
#pragma unroll
  for (int n = 0; n < DSN; ++n) Ar[n] = -__expf(A_logs[gd * DSN + n]);
  float h[DSN];
#pragma unroll
  for (int n = 0; n < DSN; ++n) h[n] = 0.f;
  for (int c = 0; c < NCK; ++c) {
    long o = ((long)ch * NCK + c) * DSN;
    float ds = Dsum[(long)ch * NCK + c];
#pragma unroll
    for (int n = 0; n < DSN; ++n) {
      Hin[o + n] = h[n];
      h[n] = __expf(ds * Ar[n]) * h[n] + Fbuf[o + n];
    }
  }
}

// ---------------- K5c: scan pass 2 — y from chained h, scatter-accumulate --
__global__ __launch_bounds__(128) void k_scan2(const _Float16* __restrict__ delta,
                                               const float* __restrict__ xconv,
                                               const float* __restrict__ xdbl,
                                               const float* __restrict__ A_logs,
                                               const float* __restrict__ Ds,
                                               const float* __restrict__ Hin,
                                               float* __restrict__ ysum) {
  int c     = blockIdx.x & (NCK - 1);
  int t     = blockIdx.x >> 4;
  int bk    = t / 3;
  int d     = (t % 3) * 128 + threadIdx.x;
  int k = bk & 3, b = bk >> 2;
  int gd = k * DI + d;
  float Ar[DSN];
#pragma unroll
  for (int n = 0; n < DSN; ++n) Ar[n] = -__expf(A_logs[gd * DSN + n]);
  float Dp = Ds[gd];
  const _Float16* dp = delta + (((long)bk * DI + d) << 12);
  const float*    up = xconv + (((long)b  * DI + d) << 12);
  const float*    bc = xdbl  + (((long)bk * CDBL + DTR) << 12);  // B,C rows
  float* yb = ysum + (long)b * LL * DI + d;
  long ch = (long)bk * DI + d;
  __shared__ float sBC[2 * DSN];
  float h[DSN];
#pragma unroll
  for (int n = 0; n < DSN; ++n) h[n] = Hin[(ch * NCK + c) * DSN + n];
  for (int l = c * CS; l < (c + 1) * CS; ++l) {
    if (threadIdx.x < 2 * DSN) sBC[threadIdx.x] = bc[((long)threadIdx.x << 12) + l];
    __syncthreads();
    float dl = (float)dp[l];
    int   s  = smap(k, l);
    float u  = up[s];
    float du = dl * u;
    float y  = 0.f;
#pragma unroll
    for (int n = 0; n < DSN; ++n) {
      h[n] = __expf(dl * Ar[n]) * h[n] + du * sBC[n];
      y   += h[n] * sBC[DSN + n];
    }
    atomicAdd(yb + (long)s * DI, y + u * Dp);     // 4 direction contributions
    __syncthreads();
  }
}

// ---------------- K6a: LayerNorm over 384 + SiLU(z) gate (1 wave / row) ----
__global__ __launch_bounds__(256) void k_lngate(const float* __restrict__ ysum,
                                                const float* __restrict__ z,
                                                const float* __restrict__ gamma,
                                                const float* __restrict__ beta,
                                                float* __restrict__ yg) {
  int lane = threadIdx.x & 31;
  int row  = blockIdx.x * 8 + (threadIdx.x >> 5);
  const float* yr = ysum + (long)row * DI;
  const float* zr = z    + (long)row * DI;
  float vals[12];
  float s1 = 0.f, s2 = 0.f;
#pragma unroll
  for (int i = 0; i < 12; ++i) {
    float v = yr[lane + i * 32];
    vals[i] = v; s1 += v; s2 += v * v;
  }
#pragma unroll
  for (int off = 16; off > 0; off >>= 1) {        // wave32 reduction
    s1 += __shfl_xor(s1, off, 32);
    s2 += __shfl_xor(s2, off, 32);
  }
  float mu  = s1 * (1.f / DI);
  float var = s2 * (1.f / DI) - mu * mu;
  float rs  = rsqrtf(var + 1e-5f);
#pragma unroll
  for (int i = 0; i < 12; ++i) {
    int   dd = lane + i * 32;
    float g  = (vals[i] - mu) * rs * gamma[dd] + beta[dd];
    float zv = zr[dd];
    g *= zv / (1.f + __expf(-zv));                // * SiLU(z)
    yg[(long)row * DI + dd] = g;
  }
}

// ---------------- K6b: out_proj GEMM out[m][c] = sum_d yg[m][d]*Wo[c][d] ---
// Wave strip: 16 x 64 (one A fragment shared by 4 WMMAs; N=192 = 3 strips).
__global__ __launch_bounds__(256) void k_outproj(const float* __restrict__ yg,
                                                 const float* __restrict__ wo,
                                                 float* __restrict__ out) {
  int lane  = threadIdx.x & 31;
  int g     = blockIdx.x * 8 + (threadIdx.x >> 5);
  int mtile = (g / 3) << 4;
  int nbase = (g % 3) << 6;
  const float* pa = yg + (long)mtile * DI;
  const float* pb = wo + (long)nbase * DI;
  v8f acc[4] = {};
#pragma unroll
  for (int kk = 0; kk < DI; kk += 32) {
    __builtin_prefetch(pa + (long)(lane & 15) * DI + kk + 64, 0, 1);
    v16h a = frag_ld_rowmajor(pa + kk, DI, lane);
#pragma unroll
    for (int j = 0; j < 4; ++j) {
      v16h b = frag_ld_rowmajor(pb + (long)(j << 4) * DI + kk, DI, lane);
      acc[j] = __builtin_amdgcn_wmma_f32_16x16x32_f16(false, a, false, b,
                                                      (short)0, acc[j], false, false);
    }
  }
  int col = lane & 15, rb = (lane >> 4) << 3;
#pragma unroll
  for (int j = 0; j < 4; ++j)
#pragma unroll
    for (int r = 0; r < 8; ++r) {
      int m = mtile + rb + r;
      out[(long)m * DM + nbase + (j << 4) + col] = acc[j][r];
    }
}

// ---------------------------------------------------------------------------
extern "C" void kernel_launch(void* const* d_in, const int* in_sizes, int n_in,
                              void* d_out, int out_size, void* d_ws, size_t ws_size,
                              hipStream_t stream) {
  (void)in_sizes; (void)n_in; (void)out_size; (void)ws_size;
  const float* x    = (const float*)d_in[0];
  const float* wip  = (const float*)d_in[1];
  const float* cw   = (const float*)d_in[2];
  const float* cb   = (const float*)d_in[3];
  const float* xpw  = (const float*)d_in[4];
  const float* dtw  = (const float*)d_in[5];
  const float* dtb  = (const float*)d_in[6];
  const float* alog = (const float*)d_in[7];
  const float* dss  = (const float*)d_in[8];
  const float* gam  = (const float*)d_in[9];
  const float* bet  = (const float*)d_in[10];
  const float* wo   = (const float*)d_in[11];
  float* out = (float*)d_out;

  char* ws = (char*)d_ws;
  const size_t SZ_BDL = (size_t)B0 * DI * LL * sizeof(float);        // ~25 MB
  const size_t SZ_DBL = (size_t)B0 * KD * CDBL * LL * sizeof(float); // ~33 MB
  const size_t SZ_DLT = (size_t)B0 * KD * DI * LL * sizeof(_Float16);// ~50 MB
  const size_t SZ_F   = (size_t)NCH * NCK * DSN * sizeof(float);     // ~6 MB
  float*    xc_pre = (float*)(ws);
  float*    xconv  = (float*)(ws + SZ_BDL);
  float*    z      = (float*)(ws + 2 * SZ_BDL);
  float*    xdbl   = (float*)(ws + 3 * SZ_BDL);
  _Float16* delta  = (_Float16*)(ws + 3 * SZ_BDL + SZ_DBL);
  float*    Fbuf   = (float*)(ws + 3 * SZ_BDL + SZ_DBL + SZ_DLT);
  float*    Hin    = (float*)(ws + 3 * SZ_BDL + SZ_DBL + SZ_DLT + SZ_F);
  float*    Dsum   = (float*)(ws + 3 * SZ_BDL + SZ_DBL + SZ_DLT + 2 * SZ_F);
  float*    ysum   = xc_pre;           // dead after conv -> reuse
  float*    yg     = (float*)delta;    // dead after scan -> reuse

  k_inproj <<<(M1 / 16) * (NXZ / 64) / 8, 256, 0, stream>>>(x, wip, xc_pre, z);
  k_conv   <<<(B0 * DI * LL) / 256,       256, 0, stream>>>(xc_pre, cw, cb, xconv);
  (void)hipMemsetAsync(ysum, 0, SZ_BDL, stream);
  k_xdbl   <<<(B0 * KD) * (CDBL / 64) * (LL / 16) / 8, 256, 0, stream>>>(xconv, xpw, xdbl);
  k_delta  <<<(B0 * KD) * (DI   / 64) * (LL / 16) / 8, 256, 0, stream>>>(xdbl, dtw, dtb, delta);
  k_scan1  <<<B0 * KD * 3 * NCK, 128, 0, stream>>>(delta, xconv, xdbl, alog, Fbuf, Dsum);
  k_chain  <<<NCH / 256, 256, 0, stream>>>(Fbuf, Dsum, alog, Hin);
  k_scan2  <<<B0 * KD * 3 * NCK, 128, 0, stream>>>(delta, xconv, xdbl, alog, dss, Hin, ysum);
  k_lngate <<<M1 / 8, 256, 0, stream>>>(ysum, z, gam, bet, yg);
  k_outproj<<<(M1 / 16) * (DM / 64) / 8, 256, 0, stream>>>(yg, wo, out);
}
